// EnergyTransformer_84043920048751
// MI455X (gfx1250) — matrix operand order repaired and checked by reading, not compiled, not run on hardware
//
#include <hip/hip_runtime.h>
#include <hip/hip_bf16.h>

// ---------------------------------------------------------------------------
// EnergyTransformer forward for MI455X (gfx1250, wave32)
//  - bf16 WMMA (v_wmma_f32_16x16x32_bf16) for all matmul-shaped ops
//  - global_load_async_to_lds_b128 + s_wait_asynccnt double-buffered staging
// ---------------------------------------------------------------------------
constexpr int B_   = 8;
constexpr int N_   = 255;
constexpr int T_   = 256;
constexpr int D_   = 768;
constexpr int H_   = 12;
constexpr int HD_  = 64;
constexpr int HID_ = 3072;
constexpr float ALPHA_ = 0.1f;
constexpr float BETA_  = 0.125f;   // 1/sqrt(64)
constexpr float EPS_   = 1e-5f;

typedef __attribute__((ext_vector_type(16))) __bf16 v16bf;
typedef __attribute__((ext_vector_type(8)))  float  v8f;

struct alignas(16) B16 { unsigned int w[4]; };
union FragBF { v16bf v; B16 q[2]; unsigned short s[16]; };
union U8x16  { B16 q; unsigned short s[8]; };

__device__ __forceinline__ unsigned short f2bf(float f) {
    unsigned int u = __float_as_uint(f);
    unsigned int r = u + 0x7FFFu + ((u >> 16) & 1u);   // round-to-nearest-even
    return (unsigned short)(r >> 16);
}

template<int Nw>
__device__ __forceinline__ void wait_async() {
    asm volatile("s_wait_asynccnt %0" :: "n"(Nw));
}

__device__ __forceinline__ void async_b128(unsigned lds_off, const void* gaddr) {
    asm volatile("global_load_async_to_lds_b128 %0, %1, off"
                 :: "v"(lds_off), "v"((unsigned long long)(uintptr_t)gaddr) : "memory");
}
__device__ __forceinline__ void async_b128_o16(unsigned lds_off, const void* gaddr) {
    asm volatile("global_load_async_to_lds_b128 %0, %1, off offset:16"
                 :: "v"(lds_off), "v"((unsigned long long)(uintptr_t)gaddr) : "memory");
}
__device__ __forceinline__ void async_b128_o32(unsigned lds_off, const void* gaddr) {
    asm volatile("global_load_async_to_lds_b128 %0, %1, off offset:32"
                 :: "v"(lds_off), "v"((unsigned long long)(uintptr_t)gaddr) : "memory");
}
__device__ __forceinline__ void async_b128_o48(unsigned lds_off, const void* gaddr) {
    asm volatile("global_load_async_to_lds_b128 %0, %1, off offset:48"
                 :: "v"(lds_off), "v"((unsigned long long)(uintptr_t)gaddr) : "memory");
}

enum { O_F32S = 0, O_F32A = 1, O_BF = 2, O_BFRELU = 3 };

// ---------------------------------------------------------------------------
// Batched bf16 GEMM: C = A_eff(256 x K) @ B_eff(K x N), f32 accumulate.
// Block tile 256x128 (grid.y==1 since M==256 everywhere), BK=32,
// 256 threads = 8 waves (4 M x 2 N), wave tile 64x64 = 16 WMMA accumulators.
// Double-buffered LDS; non-transposed operands stream via async-to-LDS.
// Requires: M==256, K%32==0, N%16==0 (true for all calls).
// ---------------------------------------------------------------------------
template<bool TrA, bool TrB, int OUT>
__global__ __launch_bounds__(256) void gemm_tpl(
    const unsigned short* __restrict__ A, const unsigned short* __restrict__ B,
    float* __restrict__ Cf, unsigned short* __restrict__ Cb,
    int N, int K, int lda, int ldb, int ldc,
    long long sA1, long long sA2, long long sB1, long long sB2,
    long long sC1, long long sC2, int nb2)
{
    constexpr int APT = (TrA ? 0 : 4) + (TrB ? 0 : 2);   // asyncs issued per tile
    __shared__ alignas(16) unsigned short lA[2][256 * 40];   // [m][k] stride 40
    __shared__ alignas(16) unsigned short lB[2][32 * 136];   // [k][n] stride 136

    const int z  = blockIdx.z;
    const int b1 = z / nb2, b2 = z - b1 * nb2;
    const unsigned short* Ab = A + b1 * sA1 + b2 * sA2;
    const unsigned short* Bb = B + b1 * sB1 + b2 * sB2;
    const long long cOff = b1 * sC1 + b2 * sC2;

    const int n0  = blockIdx.x * 128;
    const int tid = threadIdx.x;
    const int lane = tid & 31;
    const int w = tid >> 5;
    const int wm = (w >> 1) * 64;
    const int wn = (w & 1) * 64;

    auto stage = [&](int kt, int buf) {
        const int k0 = kt << 5;
        // ---- A tile [256][32] ----
        if constexpr (!TrA) {
            // one full row (32 bf16 = 64B) per thread, async straight to LDS
            unsigned lo = (unsigned)(uintptr_t)&lA[buf][tid * 40];
            const void* ga = &Ab[(long long)tid * lda + k0];
            async_b128(lo, ga); async_b128_o16(lo, ga);
            async_b128_o32(lo, ga); async_b128_o48(lo, ga);
        } else {
            // A stored [K][256] (A^T): vector-load rows of K, scatter-transpose
            const int kk = tid >> 3, seg = (tid & 7) << 5;
            const B16* src = (const B16*)&Ab[(long long)(k0 + kk) * lda + seg];
            U8x16 u[4];
            u[0].q = src[0]; u[1].q = src[1]; u[2].q = src[2]; u[3].q = src[3];
#pragma unroll
            for (int c = 0; c < 4; ++c)
#pragma unroll
                for (int j = 0; j < 8; ++j)
                    lA[buf][(seg + c * 8 + j) * 40 + kk] = u[c].s[j];
        }
        // ---- B tile [32][128] ----
        if constexpr (!TrB) {
            const int kk = tid >> 3, seg = (tid & 7) << 4;
            int gn = n0 + seg;
            if (gn + 16 > N) gn = n0;    // clamp: harmless data for unused cols
            unsigned lo = (unsigned)(uintptr_t)&lB[buf][kk * 136 + seg];
            const void* ga = &Bb[(long long)(k0 + kk) * ldb + gn];
            async_b128(lo, ga); async_b128_o16(lo, ga);
        } else {
            // B stored [N][K] (B^T): vector-load, scatter-transpose (N%128==0 here)
            const int r = tid >> 1, half = (tid & 1) << 4;
            const B16* src = (const B16*)&Bb[(long long)(n0 + r) * ldb + k0 + half];
            U8x16 u0, u1;
            u0.q = src[0]; u1.q = src[1];
#pragma unroll
            for (int j = 0; j < 8; ++j) lB[buf][(half + j) * 136 + r] = u0.s[j];
#pragma unroll
            for (int j = 0; j < 8; ++j) lB[buf][(half + 8 + j) * 136 + r] = u1.s[j];
        }
    };

    v8f acc[4][4];
#pragma unroll
    for (int mi = 0; mi < 4; ++mi)
#pragma unroll
        for (int ni = 0; ni < 4; ++ni)
            acc[mi][ni] = v8f{0.f,0.f,0.f,0.f,0.f,0.f,0.f,0.f};

    const int nk = K >> 5;
    stage(0, 0);
    for (int kt = 0; kt < nk; ++kt) {
        const int buf = kt & 1;
        if (kt + 1 < nk) { stage(kt + 1, buf ^ 1); wait_async<APT>(); }
        else             { wait_async<0>(); }
        __syncthreads();

        // A fragments: lane=row (m), K chunks {kb..kb+7, kb+16..kb+23}
        FragBF fa[4];
#pragma unroll
        for (int mi = 0; mi < 4; ++mi) {
            const int m  = wm + mi * 16 + (lane & 15);
            const int kb = (lane < 16) ? 0 : 8;
            const unsigned short* p = &lA[buf][m * 40];
            fa[mi].q[0] = *(const B16*)&p[kb];
            fa[mi].q[1] = *(const B16*)&p[kb + 16];
        }
#pragma unroll
        for (int ni = 0; ni < 4; ++ni) {
            FragBF fb;                     // lane = N col, K chunks {kb..kb+7, kb+16..kb+23}
            const int kb = (lane < 16) ? 0 : 8;
            const unsigned short* p = &lB[buf][wn + ni * 16 + (lane & 15)];
#pragma unroll
            for (int j = 0; j < 8; ++j) {
                fb.s[j]     = p[(kb + j) * 136];
                fb.s[8 + j] = p[(kb + 16 + j) * 136];
            }
#pragma unroll
            for (int mi = 0; mi < 4; ++mi)
                acc[mi][ni] = __builtin_amdgcn_wmma_f32_16x16x32_bf16(
                    false, fa[mi].v, false, fb.v, (short)0, acc[mi][ni], false, false);
        }
        __syncthreads();
    }

    // ---- epilogue: C lane layout: N = lane&15, M rows = r + 8*(lane>=16) ----
    const int cn = lane & 15;
    const int rb = (lane >> 4) << 3;
#pragma unroll
    for (int mi = 0; mi < 4; ++mi) {
#pragma unroll
        for (int ni = 0; ni < 4; ++ni) {
            const int n = n0 + wn + ni * 16 + cn;
            if (n >= N) continue;
#pragma unroll
            for (int r = 0; r < 8; ++r) {
                const int m = wm + mi * 16 + rb + r;
                const float v = acc[mi][ni][r];
                const long long idx = cOff + (long long)m * ldc + n;
                if constexpr (OUT == O_BF)          Cb[idx] = f2bf(v);
                else if constexpr (OUT == O_BFRELU) Cb[idx] = f2bf(v > 0.f ? v : 0.f);
                else if constexpr (OUT == O_F32A)   Cf[idx] += v;
                else                                Cf[idx] = v;
            }
        }
    }
}

// ---------------------------------------------------------------------------
// Small VALU kernels
// ---------------------------------------------------------------------------
__global__ void cvt_bf16_kernel(const float* __restrict__ s, unsigned short* __restrict__ d, int n) {
    int i = blockIdx.x * 256 + threadIdx.x;
    if (i < n) d[i] = f2bf(s[i]);
}

// dst[m][c][r] = src[m][r][c]  (f32 -> bf16); R rows, C cols, nm matrices
__global__ void transpose_bf16_kernel(const float* __restrict__ s, unsigned short* __restrict__ d,
                                      int R, int C, int nm) {
    long long i = (long long)blockIdx.x * 256 + threadIdx.x;
    long long per = (long long)R * C;
    if (i >= per * nm) return;
    int m  = (int)(i / per);
    int rc = (int)(i - (long long)m * per);
    int r = rc / C, c = rc - r * C;
    d[(long long)m * per + (long long)c * R + r] = f2bf(s[i]);
}

// x[b,t,:] = (t==0 ? cls : Xin[b,t-1]@W_enc + b_enc) + P[b,t]@W_pos + b_pos
__global__ void encode_kernel(const float* __restrict__ X, const float* __restrict__ P,
                              const float* __restrict__ W_enc, const float* __restrict__ b_enc,
                              const float* __restrict__ W_pos, const float* __restrict__ b_pos,
                              const float* __restrict__ cls, float* __restrict__ x)
{
    int idx = blockIdx.x * 256 + threadIdx.x;
    if (idx >= B_ * T_ * D_) return;
    int d = idx % D_;
    int t = (idx / D_) % T_;
    int b = idx / (D_ * T_);
    float acc;
    if (t == 0) {
        acc = cls[d];
    } else {
        acc = b_enc[d];
        const float* xr = &X[((long long)b * N_ + (t - 1)) * 64];
#pragma unroll 8
        for (int k = 0; k < 64; ++k) acc += xr[k] * W_enc[k * D_ + d];
    }
    float p = b_pos[d];
    const float* pr = &P[((long long)b * T_ + t) * 64];
#pragma unroll 8
    for (int k = 0; k < 64; ++k) p += pr[k] * W_pos[k * D_ + d];
    x[idx] = acc + p;
}

// edges = conv3x3(gram) * (A@W_adj + b_adj); corr[b,h,q,k] = edges; ahat = edges@W_ah
__global__ void conv_edges_kernel(const float* __restrict__ gram, const float* __restrict__ Aadj,
                                  const float* __restrict__ W_adj, const float* __restrict__ b_adj,
                                  const float* __restrict__ convk, const float* __restrict__ W_ah,
                                  float* __restrict__ corr, float* __restrict__ out_ah)
{
    long long idx = (long long)blockIdx.x * 256 + threadIdx.x;
    if (idx >= (long long)B_ * T_ * T_) return;
    int k = idx % T_;
    int q = (idx / T_) % T_;
    int b = idx / (T_ * T_);
    const float* g = &gram[(long long)b * T_ * T_];
    float tap[9];
#pragma unroll
    for (int dy = 0; dy < 3; ++dy)
#pragma unroll
        for (int dx = 0; dx < 3; ++dx) {
            int qq = q + dy - 1, kk = k + dx - 1;
            tap[dy * 3 + dx] = (qq >= 0 && qq < T_ && kk >= 0 && kk < T_) ? g[qq * T_ + kk] : 0.f;
        }
    const float* ar = &Aadj[(((long long)b * T_ + q) * T_ + k) * 8];
    float av[8];
#pragma unroll
    for (int c = 0; c < 8; ++c) av[c] = ar[c];
    float ah = 0.f;
#pragma unroll
    for (int h = 0; h < H_; ++h) {
        float cv = 0.f;
#pragma unroll
        for (int i = 0; i < 9; ++i) cv += tap[i] * convk[i * H_ + h];
        float ad = b_adj[h];
#pragma unroll
        for (int c = 0; c < 8; ++c) ad += av[c] * W_adj[c * H_ + h];
        float e = cv * ad;
        corr[(((long long)b * H_ + h) * T_ + q) * T_ + k] = e;
        ah += e * W_ah[h];
    }
    out_ah[(long long)b * T_ * T_ + q * T_ + k] = ah;
}

// g = gamma*(x-mu)*rsqrt(var+eps) + delta  -> bf16 (one workgroup per row)
__global__ void layernorm_kernel(const float* __restrict__ x, unsigned short* __restrict__ gbf,
                                 const float* __restrict__ gamma, const float* __restrict__ delta, int blk)
{
    __shared__ float red[256];
    const long long row = blockIdx.x;             // b*T + t
    const float* xr = &x[row * D_];
    const int tid = threadIdx.x;
    float v0 = xr[tid], v1 = xr[tid + 256], v2 = xr[tid + 512];
    red[tid] = v0 + v1 + v2;
    __syncthreads();
    for (int o = 128; o > 0; o >>= 1) { if (tid < o) red[tid] += red[tid + o]; __syncthreads(); }
    float mu = red[0] * (1.f / D_);
    __syncthreads();
    float c0 = v0 - mu, c1 = v1 - mu, c2 = v2 - mu;
    red[tid] = c0 * c0 + c1 * c1 + c2 * c2;
    __syncthreads();
    for (int o = 128; o > 0; o >>= 1) { if (tid < o) red[tid] += red[tid + o]; __syncthreads(); }
    float rs = rsqrtf(red[0] * (1.f / D_) + EPS_);
    float ga = gamma[blk];
    const float* de = &delta[blk * D_];
    gbf[row * D_ + tid]       = f2bf(ga * c0 * rs + de[tid]);
    gbf[row * D_ + tid + 256] = f2bf(ga * c1 * rs + de[tid + 256]);
    gbf[row * D_ + tid + 512] = f2bf(ga * c2 * rs + de[tid + 512]);
}

// Aattn = softmax_k(BETA * logits*corr) * corr   (one workgroup per (b,h,q) row)
__global__ void softmax_kernel(const float* __restrict__ logits, const float* __restrict__ corr,
                               unsigned short* __restrict__ Aat)
{
    __shared__ float red[256];
    const long long row = blockIdx.x;             // ((b*H+h)*T + q)
    const int k = threadIdx.x;
    float c = corr[row * T_ + k];
    float l = BETA_ * logits[row * T_ + k] * c;
    red[k] = l;
    __syncthreads();
    for (int o = 128; o > 0; o >>= 1) { if (k < o) red[k] = fmaxf(red[k], red[k + o]); __syncthreads(); }
    float mx = red[0];
    __syncthreads();
    float e = __expf(l - mx);
    red[k] = e;
    __syncthreads();
    for (int o = 128; o > 0; o >>= 1) { if (k < o) red[k] += red[k + o]; __syncthreads(); }
    float inv = 1.f / red[0];
    Aat[row * T_ + k] = f2bf(e * inv * c);
}

__global__ void update_kernel(float* __restrict__ x, const float* __restrict__ g, int n) {
    int i = blockIdx.x * 256 + threadIdx.x;
    if (i < n) x[i] += ALPHA_ * g[i];
}

// y = x@W_dec + b_dec; scatter CLS vs tokens into d_out (one workgroup per row)
__global__ void decode_kernel(const float* __restrict__ x, const float* __restrict__ W_dec,
                              const float* __restrict__ b_dec, float* __restrict__ out)
{
    __shared__ float xr[D_];
    const int row = blockIdx.x;                   // b*T + t
    const int b = row >> 8, t = row & 255;
    const int tid = threadIdx.x;
    xr[tid]       = x[(long long)row * D_ + tid];
    xr[tid + 256] = x[(long long)row * D_ + tid + 256];
    xr[tid + 512] = x[(long long)row * D_ + tid + 512];
    __syncthreads();
    if (tid < 64) {
        float acc = b_dec[tid];
        for (int k = 0; k < D_; ++k) acc += xr[k] * W_dec[k * 64 + tid];
        if (t == 0) out[(long long)b * 64 + tid] = acc;
        else out[(long long)B_ * 64 + ((long long)b * N_ + (t - 1)) * 64 + tid] = acc;
    }
}

// ---------------------------------------------------------------------------
extern "C" void kernel_launch(void* const* d_in, const int* in_sizes, int n_in,
                              void* d_out, int out_size, void* d_ws, size_t ws_size,
                              hipStream_t stream)
{
    const float* X     = (const float*)d_in[0];
    const float* Aadj  = (const float*)d_in[1];
    const float* P     = (const float*)d_in[2];
    const float* W_enc = (const float*)d_in[3];
    const float* b_enc = (const float*)d_in[4];
    const float* W_pos = (const float*)d_in[5];
    const float* b_pos = (const float*)d_in[6];
    const float* cls   = (const float*)d_in[7];
    const float* W_adj = (const float*)d_in[8];
    const float* b_adj = (const float*)d_in[9];
    const float* convk = (const float*)d_in[10];
    const float* gamma = (const float*)d_in[11];
    const float* delta = (const float*)d_in[12];
    const float* Wq    = (const float*)d_in[13];
    const float* Wk    = (const float*)d_in[14];
    const float* Whn   = (const float*)d_in[15];
    const float* W_dec = (const float*)d_in[16];
    const float* b_dec = (const float*)d_in[17];
    const float* W_ah  = (const float*)d_in[18];

    float* out    = (float*)d_out;
    float* out_ah = out + (size_t)B_ * T_ * 64;        // after cls(B*64) + xs(B*255*64)

    size_t off = 0;
    auto carve = [&](size_t bytes) -> void* {
        void* p = (char*)d_ws + off;
        off += (bytes + 255) & ~(size_t)255;
        return p;
    };
    const size_t BTD = (size_t)B_ * T_ * D_;
    float*          x      = (float*)carve(BTD * 4);
    float*          grads  = (float*)carve(BTD * 4);
    float*          gram   = (float*)carve((size_t)B_ * T_ * T_ * 4);
    float*          corr   = (float*)carve((size_t)B_ * H_ * T_ * T_ * 4);
    float*          logit  = (float*)carve((size_t)B_ * H_ * T_ * T_ * 4);
    unsigned short* Aat    = (unsigned short*)carve((size_t)B_ * H_ * T_ * T_ * 2);
    unsigned short* gbf    = (unsigned short*)carve(BTD * 2);
    unsigned short* QKbf   = (unsigned short*)carve(2 * BTD * 2);   // [qk][B][T][D]
    unsigned short* dQbf   = (unsigned short*)carve(BTD * 2);
    unsigned short* dKbf   = (unsigned short*)carve(BTD * 2);
    unsigned short* Hact   = (unsigned short*)carve((size_t)B_ * T_ * HID_ * 2);
    unsigned short* Wqkf   = (unsigned short*)carve((size_t)4 * D_ * D_ * 2);   // [qk][blk][hz][d]
    unsigned short* WqkTf  = (unsigned short*)carve((size_t)4 * D_ * D_ * 2);   // [qk][blk][d][hz]
    unsigned short* Whnf   = (unsigned short*)carve((size_t)2 * D_ * HID_ * 2); // [blk][d][hid]
    unsigned short* WhnTf  = (unsigned short*)carve((size_t)2 * D_ * HID_ * 2); // [blk][hid][d]

    unsigned short* Qbf = QKbf;
    unsigned short* Kbf = QKbf + BTD;

    // GEMM dispatch (M is always 256)
    auto gemm = [&](int ta, int tb, int outm,
                    const unsigned short* A, const unsigned short* Bm,
                    float* Cf, unsigned short* Cb, int N, int K,
                    int lda, int ldb, int ldc,
                    long long sA1, long long sA2, long long sB1, long long sB2,
                    long long sC1, long long sC2, int nb1, int nb2) {
        dim3 g((N + 127) / 128, 1, nb1 * nb2);
#define GEMM_CASE(TAv, TBv, Ov) \
        gemm_tpl<TAv, TBv, Ov><<<g, 256, 0, stream>>>(A, Bm, Cf, Cb, N, K, lda, ldb, ldc, \
                                                      sA1, sA2, sB1, sB2, sC1, sC2, nb2)
        if (!ta && !tb) {
            if      (outm == O_F32S)   GEMM_CASE(false, false, O_F32S);
            else if (outm == O_F32A)   GEMM_CASE(false, false, O_F32A);
            else if (outm == O_BF)     GEMM_CASE(false, false, O_BF);
            else                       GEMM_CASE(false, false, O_BFRELU);
        } else if (!ta && tb) {
            if      (outm == O_F32S)   GEMM_CASE(false, true, O_F32S);
            else                       GEMM_CASE(false, true, O_BF);
        } else {
            GEMM_CASE(true, false, O_BF);
        }
#undef GEMM_CASE
    };
    auto cvt = [&](const float* s, unsigned short* d, size_t n) {
        cvt_bf16_kernel<<<(unsigned)((n + 255) / 256), 256, 0, stream>>>(s, d, (int)n);
    };

    const long long TD   = (long long)T_ * D_;
    const long long TT   = (long long)T_ * T_;
    const long long HTT  = (long long)H_ * T_ * T_;
    const long long THID = (long long)T_ * HID_;
    const long long DD   = (long long)D_ * D_;

    // ---- setup: bf16 weights (plain + transposed), encoder, gram, edges ----
    cvt(Wq,  Wqkf,            (size_t)2 * DD);
    cvt(Wk,  Wqkf + 2 * DD,   (size_t)2 * DD);
    cvt(Whn, Whnf,            (size_t)2 * D_ * HID_);
    {
        unsigned nb = (unsigned)((2 * DD + 255) / 256);
        transpose_bf16_kernel<<<nb, 256, 0, stream>>>(Wq, WqkTf,          D_, D_, 2);
        transpose_bf16_kernel<<<nb, 256, 0, stream>>>(Wk, WqkTf + 2 * DD, D_, D_, 2);
        unsigned nb2 = (unsigned)(((long long)2 * D_ * HID_ + 255) / 256);
        transpose_bf16_kernel<<<nb2, 256, 0, stream>>>(Whn, WhnTf, D_, HID_, 2);
    }

    encode_kernel<<<(unsigned)((BTD + 255) / 256), 256, 0, stream>>>(
        X, P, W_enc, b_enc, W_pos, b_pos, cls, x);

    cvt(x, gbf, BTD);   // x -> bf16 for gram
    // gram[b] = x[b] @ x[b]^T
    gemm(0, 1, O_F32S, gbf, gbf, gram, nullptr, T_, D_, D_, D_, T_,
         TD, 0, TD, 0, TT, 0, B_, 1);

    conv_edges_kernel<<<(unsigned)(((long long)B_ * T_ * T_ + 255) / 256), 256, 0, stream>>>(
        gram, Aadj, W_adj, b_adj, convk, W_ah, corr, out_ah);

    // ---- 2 blocks x 12 energy-descent steps ----
    for (int blk = 0; blk < 2; ++blk) {
        const unsigned short* Wqb   = Wqkf  + (size_t)blk * DD;            // [hz][d]
        const unsigned short* Wkb   = Wqkf  + (size_t)(2 + blk) * DD;
        const unsigned short* WqkTb = WqkTf + (size_t)blk * DD;            // [d][hz], qk stride 2*DD
        const unsigned short* Whnb  = Whnf  + (size_t)blk * D_ * HID_;     // [d][hid]
        const unsigned short* WhnTb = WhnTf + (size_t)blk * D_ * HID_;     // [hid][d]
        for (int it = 0; it < 12; ++it) {
            layernorm_kernel<<<B_ * T_, 256, 0, stream>>>(x, gbf, gamma, delta, blk);

            // Q,K = g @ Wq^T / g @ Wk^T  (one launch: batch = (b, qk))
            gemm(0, 0, O_BF, gbf, WqkTb, nullptr, QKbf, D_, D_, D_, D_, D_,
                 TD, 0, 0, 2 * DD, TD, (long long)B_ * TD, B_, 2);

            // logits[b,h] = Q_h @ K_h^T
            gemm(0, 1, O_F32S, Qbf, Kbf, logit, nullptr, T_, HD_, D_, D_, T_,
                 TD, HD_, TD, HD_, HTT, TT, B_, H_);

            softmax_kernel<<<B_ * H_ * T_, 256, 0, stream>>>(logit, corr, Aat);

            // dQ_h = Aattn @ K_h ; dK_h = Aattn^T @ Q_h
            gemm(0, 0, O_BF, Aat, Kbf, nullptr, dQbf, HD_, T_, T_, D_, D_,
                 HTT, TT, TD, HD_, TD, HD_, B_, H_);
            gemm(1, 0, O_BF, Aat, Qbf, nullptr, dKbf, HD_, T_, T_, D_, D_,
                 HTT, TT, TD, HD_, TD, HD_, B_, H_);

            // grads = dQ @ Wq + dK @ Wk
            gemm(0, 0, O_F32S, dQbf, Wqb, grads, nullptr, D_, D_, D_, D_, D_,
                 TD, 0, 0, 0, TD, 0, B_, 1);
            gemm(0, 0, O_F32A, dKbf, Wkb, grads, nullptr, D_, D_, D_, D_, D_,
                 TD, 0, 0, 0, TD, 0, B_, 1);

            // Hact = relu(g @ Whn) ; grads += Hact @ Whn^T
            gemm(0, 0, O_BFRELU, gbf, Whnb, nullptr, Hact, HID_, D_, D_, HID_, HID_,
                 TD, 0, 0, 0, THID, 0, B_, 1);
            gemm(0, 0, O_F32A, Hact, WhnTb, grads, nullptr, D_, HID_, HID_, D_, D_,
                 THID, 0, 0, 0, TD, 0, B_, 1);

            // x <- x - ALPHA * dE/dg = x + ALPHA * grads
            update_kernel<<<(unsigned)((BTD + 255) / 256), 256, 0, stream>>>(x, grads, (int)BTD);
        }
    }

    // ---- decode: y = x @ W_dec + b_dec -> (cls, xs) ----
    decode_kernel<<<B_ * T_, 256, 0, stream>>>(x, W_dec, b_dec, out);
}
